// VQEmbedding_63024350102040
// MI455X (gfx1250) — matrix operand order repaired and checked by reading, not compiled, not run on hardware
//
#include <hip/hip_runtime.h>

typedef __attribute__((ext_vector_type(16))) _Float16 v16h;
typedef __attribute__((ext_vector_type(8)))  _Float16 v8h;
typedef __attribute__((ext_vector_type(8)))  float    v8f;

#define KCODES 512
#define DDIM 64
#define CB_STRIDE 72   // padded f16 row stride in LDS (36 dwords -> conflict-free B loads)
#define WAVES_PER_BLOCK 8
#define ROWS_PER_TILE 16
#define ROWS_PER_PASS (WAVES_PER_BLOCK * ROWS_PER_TILE)  // 128
#define PASSES 4
#define ROWS_PER_BLOCK (ROWS_PER_PASS * PASSES)          // 512

__device__ __forceinline__ unsigned int mono_f32(float f) {
  // order-preserving float -> uint mapping
  unsigned int b = __float_as_uint(f);
  unsigned int m = ((unsigned int)((int)b >> 31)) | 0x80000000u;
  return b ^ m;
}

__global__ __launch_bounds__(256) void vq_argmin_kernel(
    const float* __restrict__ z, const float* __restrict__ cb,
    float* __restrict__ out, int nRows)
{
  extern __shared__ char smem[];
  _Float16* cbLds = (_Float16*)smem;                                        // [512][72] f16
  float*    hLds  = (float*)(smem + KCODES * CB_STRIDE * sizeof(_Float16)); // [512] = 0.5*||e||^2

  // ---- Stage codebook into LDS (f16) + per-codeword half-norms (f32) ----
  for (int n = threadIdx.x; n < KCODES; n += blockDim.x) {
    const float* cr = cb + n * DDIM;
    _Float16* dst = cbLds + n * CB_STRIDE;
    float ss = 0.f;
#pragma unroll
    for (int d = 0; d < DDIM; d += 4) {
      float4 v = *(const float4*)(cr + d);
      ss += v.x * v.x + v.y * v.y + v.z * v.z + v.w * v.w;
      dst[d + 0] = (_Float16)v.x;
      dst[d + 1] = (_Float16)v.y;
      dst[d + 2] = (_Float16)v.z;
      dst[d + 3] = (_Float16)v.w;
    }
    hLds[n] = 0.5f * ss;
  }
  __syncthreads();

  const int lane  = threadIdx.x & 31;
  const int wave  = threadIdx.x >> 5;
  const int lhalf = lane >> 4;     // 0: lanes 0-15, 1: lanes 16-31
  const int l16   = lane & 15;

  for (int pass = 0; pass < PASSES; ++pass) {
    int rowBase = (int)blockIdx.x * ROWS_PER_BLOCK + pass * ROWS_PER_PASS
                  + wave * ROWS_PER_TILE;
    if (rowBase >= nRows) break;   // wave-uniform

    // ---- A fragments: 16 rows of z, D=64 -> two 16x32 f16 fragments ----
    // ISA layout: lane<16 = row l16, K in {0..7, 16..23}; lane>=16 = row l16, K in {8..15, 24..31}
    int arow = rowBase + l16;
    if (arow >= nRows) arow = nRows - 1;
    const float* zr = z + (size_t)arow * DDIM;
    v16h afrag[2];
#pragma unroll
    for (int s = 0; s < 2; ++s) {
      int dlo = 32 * s + (lhalf ? 8 : 0);
      int dhi = dlo + 16;
      float4 x0 = *(const float4*)(zr + dlo);
      float4 x1 = *(const float4*)(zr + dlo + 4);
      float4 y0 = *(const float4*)(zr + dhi);
      float4 y1 = *(const float4*)(zr + dhi + 4);
      v16h a;
      a[0]  = (_Float16)x0.x; a[1]  = (_Float16)x0.y; a[2]  = (_Float16)x0.z; a[3]  = (_Float16)x0.w;
      a[4]  = (_Float16)x1.x; a[5]  = (_Float16)x1.y; a[6]  = (_Float16)x1.z; a[7]  = (_Float16)x1.w;
      a[8]  = (_Float16)y0.x; a[9]  = (_Float16)y0.y; a[10] = (_Float16)y0.z; a[11] = (_Float16)y0.w;
      a[12] = (_Float16)y1.x; a[13] = (_Float16)y1.y; a[14] = (_Float16)y1.z; a[15] = (_Float16)y1.w;
      afrag[s] = a;
    }

    float bestVal[8];
    int   bestIdx[8];
#pragma unroll
    for (int r = 0; r < 8; ++r) { bestVal[r] = -3.4e38f; bestIdx[r] = 0; }

    // ---- Sweep 512 codewords, two 16-wide column tiles per iteration ----
    // Two independent WMMA accumulator chains fill each other's hazard slots.
    for (int c = 0; c < KCODES / 16; c += 2) {
      int n0 = c * 16 + l16;
      int n1 = n0 + 16;
      const _Float16* crow0 = cbLds + n0 * CB_STRIDE;
      const _Float16* crow1 = cbLds + n1 * CB_STRIDE;
      int kb0 = (lhalf ? 8 : 0);        // s = 0
      int kb1 = 32 + (lhalf ? 8 : 0);   // s = 1

      v16h b00, b01, b10, b11;
      {
        const v8h* p = (const v8h*)(crow0 + kb0);
        v8h lo = p[0], hi = p[2];
#pragma unroll
        for (int i = 0; i < 8; ++i) { b00[i] = lo[i]; b00[8 + i] = hi[i]; }
      }
      {
        const v8h* p = (const v8h*)(crow0 + kb1);
        v8h lo = p[0], hi = p[2];
#pragma unroll
        for (int i = 0; i < 8; ++i) { b01[i] = lo[i]; b01[8 + i] = hi[i]; }
      }
      {
        const v8h* p = (const v8h*)(crow1 + kb0);
        v8h lo = p[0], hi = p[2];
#pragma unroll
        for (int i = 0; i < 8; ++i) { b10[i] = lo[i]; b10[8 + i] = hi[i]; }
      }
      {
        const v8h* p = (const v8h*)(crow1 + kb1);
        v8h lo = p[0], hi = p[2];
#pragma unroll
        for (int i = 0; i < 8; ++i) { b11[i] = lo[i]; b11[8 + i] = hi[i]; }
      }

      float h0 = hLds[n0];
      float h1 = hLds[n1];

      v8f acc0 = {};
      v8f acc1 = {};
      acc0 = __builtin_amdgcn_wmma_f32_16x16x32_f16(false, afrag[0], false, b00,
                                                    (short)0, acc0, false, false);
      acc1 = __builtin_amdgcn_wmma_f32_16x16x32_f16(false, afrag[0], false, b10,
                                                    (short)0, acc1, false, false);
      acc0 = __builtin_amdgcn_wmma_f32_16x16x32_f16(false, afrag[1], false, b01,
                                                    (short)0, acc0, false, false);
      acc1 = __builtin_amdgcn_wmma_f32_16x16x32_f16(false, afrag[1], false, b11,
                                                    (short)0, acc1, false, false);

#pragma unroll
      for (int r = 0; r < 8; ++r) {
        float s0 = acc0[r] - h0;   // z.e - 0.5||e||^2 ; maximize
        if (s0 > bestVal[r]) { bestVal[r] = s0; bestIdx[r] = n0; }
        float s1 = acc1[r] - h1;
        if (s1 > bestVal[r]) { bestVal[r] = s1; bestIdx[r] = n1; }
      }
    }

    // ---- Branchless argmax reduction across the 16 lanes of each row ----
    // Pack (score, ~idx) into one monotone u64 key; ties pick the lower index.
    unsigned long long key[8];
#pragma unroll
    for (int r = 0; r < 8; ++r) {
      key[r] = ((unsigned long long)mono_f32(bestVal[r]) << 32)
             | (unsigned int)(~bestIdx[r]);
    }
#pragma unroll
    for (int m = 8; m >= 1; m >>= 1) {
#pragma unroll
      for (int r = 0; r < 8; ++r) {
        unsigned long long ok = __shfl_xor(key[r], m, 32);
        key[r] = (ok > key[r]) ? ok : key[r];
      }
    }

    // C/D layout: lanes 0-15 hold rows r (VGPR r), lanes 16-31 hold rows r+8
    if (l16 == 0) {
      int rbase = rowBase + lhalf * 8;
#pragma unroll
      for (int r = 0; r < 8; ++r) {
        int row = rbase + r;
        if (row < nRows) {
          unsigned int idx = ~(unsigned int)(key[r] & 0xFFFFFFFFu);
          out[row] = (float)idx;
        }
      }
    }
  }
}

extern "C" void kernel_launch(void* const* d_in, const int* in_sizes, int n_in,
                              void* d_out, int out_size, void* d_ws, size_t ws_size,
                              hipStream_t stream) {
  (void)n_in; (void)out_size; (void)d_ws; (void)ws_size;
  const float* z  = (const float*)d_in[0];
  const float* cb = (const float*)d_in[1];
  float* out = (float*)d_out;

  int nRows  = in_sizes[0] / DDIM;   // 262144
  int blocks = (nRows + ROWS_PER_BLOCK - 1) / ROWS_PER_BLOCK;  // 512
  size_t lds = (size_t)KCODES * CB_STRIDE * sizeof(_Float16) + KCODES * sizeof(float);

  vq_argmin_kernel<<<blocks, 256, lds, stream>>>(z, cb, out, nRows);
}
